// GrangerGraphEncoder_55825984913884
// MI455X (gfx1250) — compile-verified, matrix operand order checked by reading, not run on hardware
//
#include <hip/hip_runtime.h>
#include <hip/hip_bf16.h>
#include <math.h>

typedef __attribute__((ext_vector_type(16))) _Float16 v16h;
typedef __attribute__((ext_vector_type(8)))  float    v8f;
typedef __attribute__((ext_vector_type(4)))  unsigned int u32x4;
typedef __attribute__((ext_vector_type(8)))  unsigned int u32x8;

// Problem constants (from reference)
constexpr int CB = 4, CK = 8, CN = 512, CFN = 32, CFE = 4, CDE = 16, CD = 128, CH = 4, CL = 2;

#define RFL(x) ((unsigned int)__builtin_amdgcn_readfirstlane((int)(x)))

// ---------------------------------------------------------------------------
// WMMA fragment helpers (wave32, 16x16x32 f16 -> f32). ISA 7.12.2 layouts:
//  A (16xK f16): lane&15 = M row, lane>>4 selects K-half; elem i<8 -> K=kh*8+i,
//                elem i>=8 -> K=16+kh*8+(i-8)  => two contiguous 8-float runs.
//  B (Kx16 f16): lane&15 = N col, same K mapping per element (strided rows).
//  C/D (16x16 f32): 8 VGPRs; lanes 0-15 rows 0-7, lanes 16-31 rows 8-15, col=lane&15.
// ---------------------------------------------------------------------------
__device__ __forceinline__ int kmap(int i, int kh) {
    return (i < 8) ? (kh * 8 + i) : (16 + kh * 8 + (i - 8));
}

// Pack two contiguous float4 pairs (runs at +0 and +16) into a v16h fragment.
__device__ __forceinline__ v16h pack_fragment(const float* __restrict__ p) {
    float4 q0 = *(const float4*)(p);
    float4 q1 = *(const float4*)(p + 4);
    float4 q2 = *(const float4*)(p + 16);
    float4 q3 = *(const float4*)(p + 20);
    v16h a;
    a[0]  = (_Float16)q0.x; a[1]  = (_Float16)q0.y; a[2]  = (_Float16)q0.z; a[3]  = (_Float16)q0.w;
    a[4]  = (_Float16)q1.x; a[5]  = (_Float16)q1.y; a[6]  = (_Float16)q1.z; a[7]  = (_Float16)q1.w;
    a[8]  = (_Float16)q2.x; a[9]  = (_Float16)q2.y; a[10] = (_Float16)q2.z; a[11] = (_Float16)q2.w;
    a[12] = (_Float16)q3.x; a[13] = (_Float16)q3.y; a[14] = (_Float16)q3.z; a[15] = (_Float16)q3.w;
    return a;
}

// A tile: rows row_base..+15, K window [k0, k0+32). Requires k0 % 8 == 0, ld % 8 == 0.
__device__ __forceinline__ v16h load_a_tile(const float* __restrict__ A, int ld,
                                            int row_base, int k0) {
    int lane = threadIdx.x & 31;
    int r = lane & 15, kh = lane >> 4;
    return pack_fragment(A + (long)(row_base + r) * ld + k0 + kh * 8);
}

// B tile (row-major K x N source, strided per-K loads; coalesced across lanes)
__device__ __forceinline__ v16h load_b_tile(const float* __restrict__ Bm, int ld,
                                            int k0, int n0) {
    int lane = threadIdx.x & 31;
    int n = lane & 15, kh = lane >> 4;
    v16h b;
#pragma unroll
    for (int i = 0; i < 16; ++i) b[i] = (_Float16)Bm[(long)(k0 + kmap(i, kh)) * ld + n0 + n];
    return b;
}

// B tile where B[kk][n] = M[(row_base+n)*ld + k0 + kk]   (i.e. B = M^T slice)
__device__ __forceinline__ v16h load_bT_tile(const float* __restrict__ M, int ld,
                                             int row_base, int k0) {
    int lane = threadIdx.x & 31;
    int n = lane & 15, kh = lane >> 4;
    return pack_fragment(M + (long)(row_base + n) * ld + k0 + kh * 8);
}

__device__ __forceinline__ void store_c_tile(float* __restrict__ C, int ld,
                                             int m0, int n0, v8f c) {
    int lane = threadIdx.x & 31;
    int col = lane & 15;
    int rb = (lane >> 4) * 8;
#pragma unroll
    for (int j = 0; j < 8; ++j) C[(long)(m0 + rb + j) * ld + n0 + col] = c[j];
}

__device__ __forceinline__ float elu_f(float x) { return x > 0.f ? x : (expf(x) - 1.f); }
__device__ __forceinline__ float sigm_f(float x) { return 1.f / (1.f + expf(-x)); }

// Block-wide (128 thread) sum via wave32 shfl + LDS
__device__ __forceinline__ float block_sum_128(float v, float* sbuf) {
#pragma unroll
    for (int off = 16; off > 0; off >>= 1) v += __shfl_xor(v, off);
    __syncthreads();
    if ((threadIdx.x & 31) == 0) sbuf[threadIdx.x >> 5] = v;
    __syncthreads();
    return sbuf[0] + sbuf[1] + sbuf[2] + sbuf[3];
}

// ---------------------------------------------------------------------------
// Generic f32->f16 WMMA GEMM: C[M,N] = A[M,K] @ B[K,N] (+bias[N]).
// One wave computes a 16x32 strip (A fragment reused across 2 B tiles).
// grid = (M/16, N/32), block = 32. K % 32 == 0, N % 32 == 0.
// ---------------------------------------------------------------------------
__global__ void wmma_gemm_kernel(const float* __restrict__ A, const float* __restrict__ B,
                                 const float* __restrict__ bias, float* __restrict__ C,
                                 int Kd, int Nd) {
    int m0 = blockIdx.x * 16, n0 = blockIdx.y * 32;
    v8f acc0 = {}, acc1 = {};
    for (int k0 = 0; k0 < Kd; k0 += 32) {
        if (k0 + 32 < Kd)
            __builtin_prefetch(A + (long)m0 * Kd + k0 + 32, 0, 1);   // global_prefetch_b8
        v16h a  = load_a_tile(A, Kd, m0, k0);
        v16h b0 = load_b_tile(B, Nd, k0, n0);
        v16h b1 = load_b_tile(B, Nd, k0, n0 + 16);
        acc0 = __builtin_amdgcn_wmma_f32_16x16x32_f16(false, a, false, b0, (short)0, acc0, false, false);
        acc1 = __builtin_amdgcn_wmma_f32_16x16x32_f16(false, a, false, b1, (short)0, acc1, false, false);
    }
    if (bias) {
        float bv0 = bias[n0 + (threadIdx.x & 15)];
        float bv1 = bias[n0 + 16 + (threadIdx.x & 15)];
#pragma unroll
        for (int j = 0; j < 8; ++j) { acc0[j] += bv0; acc1[j] += bv1; }
    }
    store_c_tile(C, Nd, m0, n0, acc0);
    store_c_tile(C, Nd, m0, n0 + 16, acc1);
}

// h0 += time_embed[k] broadcast   (inp_b already folded into the GEMM)
__global__ void add_te_kernel(float* __restrict__ h0, const float* __restrict__ te) {
    long idx = (long)blockIdx.x * 256 + threadIdx.x;     // B*K*N*D = 2^21
    if (idx >= (long)CB * CK * CN * CD) return;
    int d = idx & (CD - 1);
    int k = (idx >> 16) & (CK - 1);                      // idx/(D*N) % K, D*N = 65536
    h0[idx] += te[k * CD + d];
}

// ---------------------------------------------------------------------------
// Node temporal conv (D->D, kernel 3, causal, last step) + LN residual.
// One 128-thread block per (b,n).
// ---------------------------------------------------------------------------
__global__ void node_conv_kernel(const float* __restrict__ h0,
                                 const float* __restrict__ w1, const float* __restrict__ b1,
                                 const float* __restrict__ w2, const float* __restrict__ b2,
                                 const float* __restrict__ g, const float* __restrict__ bb,
                                 float* __restrict__ hout) {
    __shared__ float s_in[5][CD];
    __shared__ float s_mid[3][CD];
    __shared__ float sred[4];
    int c = threadIdx.x;
    int bn = blockIdx.x;
    int b = bn >> 9, n = bn & (CN - 1);
#pragma unroll
    for (int t = 0; t < 5; ++t)
        s_in[t][c] = h0[((long)(b * CK + (3 + t)) * CN + n) * CD + c];
    __syncthreads();
#pragma unroll
    for (int tt = 0; tt < 3; ++tt) {                     // conv1 at t = 5,6,7
        float acc = b1[c];
#pragma unroll
        for (int tap = 0; tap < 3; ++tap) {
            const float* wr = w1 + (long)c * CD * 3 + tap;
            const float* xi = s_in[tt + tap];
            for (int ci = 0; ci < CD; ++ci) acc += wr[ci * 3] * xi[ci];
        }
        s_mid[tt][c] = elu_f(acc);
    }
    __syncthreads();
    float acc = b2[c];                                   // conv2 at t = 7
#pragma unroll
    for (int tap = 0; tap < 3; ++tap) {
        const float* wr = w2 + (long)c * CD * 3 + tap;
        const float* xi = s_mid[tap];
        for (int ci = 0; ci < CD; ++ci) acc += wr[ci * 3] * xi[ci];
    }
    float x = acc + s_in[4][c];                          // + h[:, -1]
    float m = block_sum_128(x, sred) * (1.f / CD);
    float d = x - m;
    float var = block_sum_128(d * d, sred) * (1.f / CD);
    hout[(long)(b * CN + n) * CD + c] = d * rsqrtf(var + 1e-5f) * g[c] + bb[c];
}

// ---------------------------------------------------------------------------
// Edge encoder: per wave, 16 edge pixels (fixed b,i; 16 consecutive j).
// Input slab staged Global->LDS with one TDM tensor_load_to_lds (2-D tile:
// 64 floats x 8 K-lines, line stride N*N*FE), waited with s_wait_tensorcnt.
// conv1 (FE*3=12 -> pad K=32, 1 WMMA/step) -> elu -> conv2 (DE*3=48 -> pad
// K=64, 2 WMMAs/step) -> softmax-attn over k -> LN -> edge_enc[b,i,j,DE].
// ---------------------------------------------------------------------------
__global__ void edge_encoder_kernel(const float* __restrict__ ef,
                                    const float* __restrict__ w1, const float* __restrict__ b1v,
                                    const float* __restrict__ w2, const float* __restrict__ b2v,
                                    const float* __restrict__ aw,
                                    const float* __restrict__ ng, const float* __restrict__ nb,
                                    float* __restrict__ out) {
    __shared__ float s_ef[CK][16 * CFE];       // [t][pixel*4+f]  (TDM destination)
    __shared__ float s_x1[CK][16][CDE];
    __shared__ float s_x2[CK][16][CDE];
    __shared__ float s_logit[16][CK];

    int lane = threadIdx.x;
    int tile = blockIdx.x;                     // B*N*N/16 = 65536 tiles
    int b = tile >> 14;                        // 16384 tiles per batch
    int rem = tile & 16383;
    int i = rem >> 5;                          // 32 j-tiles per row
    int j0 = (rem & 31) * 16;

    // ---- Tensor Data Mover: D# for a 2D tile (64 elems x 8 lines, f32) ----
    unsigned int lds_off = (unsigned int)(size_t)(&s_ef[0][0]);
    unsigned long long gaddr =
        (unsigned long long)(size_t)(ef + (((long)b * CK * CN + i) * CN + j0) * CFE);
    u32x4 g0;
    g0[0] = RFL(1u);                                          // count=1, user mode
    g0[1] = RFL(lds_off);                                     // lds_addr
    g0[2] = RFL((unsigned int)gaddr);                         // global_addr[31:0]
    g0[3] = RFL((((unsigned int)(gaddr >> 32)) & 0x01FFFFFFu) // global_addr[56:32]
                | 0x80000000u);                               // type=2 ("image")
    u32x8 g1;
    g1[0] = RFL(0x00020000u);            // wg_mask=0, data_size=2 (4B), no pad/iter
    g1[1] = RFL(64u << 16);              // tensor_dim0 = 64 (lo16)
    g1[2] = RFL(8u << 16);               // tensor_dim0 hi=0, tensor_dim1 = 8 (lo16)
    g1[3] = RFL(64u << 16);              // tensor_dim1 hi=0, tile_dim0 = 64
    g1[4] = RFL(8u);                     // tile_dim1 = 8, tile_dim2 = 0
    g1[5] = RFL((unsigned int)(CN * CN * CFE));  // tensor_dim0_stride lo32
    g1[6] = RFL(0u);                     // stride hi, tensor_dim1_stride lo16
    g1[7] = RFL(0u);
    asm volatile("tensor_load_to_lds %0, %1" :: "s"(g0), "s"(g1) : "memory");
    __builtin_amdgcn_s_wait_tensorcnt(0);
    __syncthreads();

    int r = lane & 15, kh = lane >> 4;         // r = pixel row / de col, kh = K half
    // conv1 weight fragment: B[kk][de], kk = tap*4+f (<12), padded to 32
    v16h b1;
#pragma unroll
    for (int e = 0; e < 16; ++e) {
        int kk = kmap(e, kh);
        float v = 0.f;
        if (kk < 12) { int tap = kk >> 2, f = kk & 3; v = w1[(r * CFE + f) * 3 + tap]; }
        b1[e] = (_Float16)v;
    }
    float bias1 = b1v[r];
    for (int t = 0; t < CK; ++t) {
        v16h a;
#pragma unroll
        for (int e = 0; e < 16; ++e) {
            int kk = kmap(e, kh);
            float v = 0.f;
            if (kk < 12) {
                int tap = kk >> 2, f = kk & 3, tin = t - 2 + tap;
                if (tin >= 0) v = s_ef[tin][r * CFE + f];
            }
            a[e] = (_Float16)v;
        }
        v8f c = {};
        c = __builtin_amdgcn_wmma_f32_16x16x32_f16(false, a, false, b1, (short)0, c, false, false);
        int rb = kh * 8;
#pragma unroll
        for (int j = 0; j < 8; ++j) s_x1[t][rb + j][r] = elu_f(c[j] + bias1);
    }
    __syncthreads();

    // conv2 weight fragments: B[kk][dout], kk = tap*16+din (<48), padded to 64
    v16h b2[2];
#pragma unroll
    for (int ch = 0; ch < 2; ++ch) {
#pragma unroll
        for (int e = 0; e < 16; ++e) {
            int kk = ch * 32 + kmap(e, kh);
            float v = 0.f;
            if (kk < 48) { int tap = kk >> 4, din = kk & 15; v = w2[(r * CDE + din) * 3 + tap]; }
            b2[ch][e] = (_Float16)v;
        }
    }
    float bias2 = b2v[r], awv = aw[r];
    for (int t = 0; t < CK; ++t) {
        v8f c = {};
#pragma unroll
        for (int ch = 0; ch < 2; ++ch) {
            v16h a;
#pragma unroll
            for (int e = 0; e < 16; ++e) {
                int kk = ch * 32 + kmap(e, kh);
                float v = 0.f;
                if (kk < 48) {
                    int tap = kk >> 4, din = kk & 15, tin = t - 2 + tap;
                    if (tin >= 0) v = s_x1[tin][r][din];
                }
                a[e] = (_Float16)v;
            }
            c = __builtin_amdgcn_wmma_f32_16x16x32_f16(false, a, false, b2[ch],
                                                       (short)0, c, false, false);
        }
        int rb = kh * 8;
#pragma unroll
        for (int j = 0; j < 8; ++j) {
            float x = c[j] + bias2;
            s_x2[t][rb + j][r] = x;
            float p = x * awv;                 // logits = x2 . attn_w
            p += __shfl_xor(p, 1); p += __shfl_xor(p, 2);
            p += __shfl_xor(p, 4); p += __shfl_xor(p, 8);
            if ((lane & 15) == 0) s_logit[rb + j][t] = p;
        }
    }
    __syncthreads();

    // per-pixel softmax over k, attention-pool, LN over DE=16
    int p = lane & 15;
    float mx = -1e30f;
#pragma unroll
    for (int t = 0; t < CK; ++t) mx = fmaxf(mx, s_logit[p][t]);
    float wgt[CK], se = 0.f;
#pragma unroll
    for (int t = 0; t < CK; ++t) { wgt[t] = expf(s_logit[p][t] - mx); se += wgt[t]; }
    float inv = 1.f / se;
    float pooled[8], s1 = 0.f, s2 = 0.f;
#pragma unroll
    for (int dd = 0; dd < 8; ++dd) {
        float acc = 0.f;
#pragma unroll
        for (int t = 0; t < CK; ++t) acc += wgt[t] * s_x2[t][p][kh * 8 + dd];
        pooled[dd] = acc * inv;
        s1 += pooled[dd]; s2 += pooled[dd] * pooled[dd];
    }
    s1 += __shfl_xor(s1, 16); s2 += __shfl_xor(s2, 16);
    float mean = s1 * (1.f / CDE);
    float var = s2 * (1.f / CDE) - mean * mean;
    float rstd = rsqrtf(var + 1e-5f);
    long obase = ((long)(b * CN + i) * CN + (j0 + p)) * CDE;
#pragma unroll
    for (int dd = 0; dd < 8; ++dd) {
        int de = kh * 8 + dd;
        out[obase + de] = (pooled[dd] - mean) * rstd * ng[de] + nb[de];
    }
}

// bias[l,b,h,i,j] = edge_enc[b,j,i,:] . eb_w[l,:,h] + eb_b[l,h]
__global__ void edge_bias_kernel(const float* __restrict__ enc, const float* __restrict__ ebw,
                                 const float* __restrict__ ebb, float* __restrict__ bias) {
    long idx = (long)blockIdx.x * 256 + threadIdx.x;     // L*B*H*N*N = 8388608
    if (idx >= (long)CL * CB * CH * CN * CN) return;
    int j = idx & (CN - 1); long t = idx >> 9;
    int i = t & (CN - 1); t >>= 9;
    int hh = t & (CH - 1); t >>= 2;
    int b = t & (CB - 1); int l = (int)(t >> 2);
    const float* e = enc + ((long)(b * CN + j) * CN + i) * CDE;
    const float* w = ebw + (long)l * CDE * CH;
    float acc = ebb[l * CH + hh];
#pragma unroll
    for (int de = 0; de < CDE; ++de) acc += e[de] * w[de * CH + hh];
    bias[idx] = acc;
}

// ---------------------------------------------------------------------------
// Attention core: one wave per (b, h, 16-row i-tile). Scores via WMMA into
// 32KB LDS, row softmax, then P@V via WMMA (32 j-cols per step).
// ---------------------------------------------------------------------------
__global__ void attn_kernel(const float* __restrict__ Q, const float* __restrict__ Kmat,
                            const float* __restrict__ V, const float* __restrict__ bias,
                            float* __restrict__ O) {
    __shared__ float s_S[16][CN];
    int lane = threadIdx.x;
    int blk = blockIdx.x;
    int it = blk & 31, hh = (blk >> 5) & 3, b = blk >> 7;
    int i0 = it * 16, rowb = b * CN;
    int kh = lane >> 4, nn = lane & 15, rb = kh * 8;
    const float scale = 0.17677669529663687f;            // 1/sqrt(DH=32)
    const float* bptr = bias + (long)(b * CH + hh) * CN * CN;

    v16h aq = load_a_tile(Q, CD, rowb + i0, hh * 32);
    for (int jt = 0; jt < 32; ++jt) {
        v16h bk = load_bT_tile(Kmat, CD, rowb + jt * 16, hh * 32);
        v8f s = {};
        s = __builtin_amdgcn_wmma_f32_16x16x32_f16(false, aq, false, bk, (short)0, s, false, false);
#pragma unroll
        for (int j = 0; j < 8; ++j) {
            int row = rb + j;
            s_S[row][jt * 16 + nn] = s[j] * scale + bptr[(long)(i0 + row) * CN + jt * 16 + nn];
        }
    }
    __syncthreads();

    int row = lane >> 1, seg = lane & 1;                 // 2 lanes per row
    float mx = -1e30f;
    for (int c = seg * 256; c < seg * 256 + 256; ++c) mx = fmaxf(mx, s_S[row][c]);
    mx = fmaxf(mx, __shfl_xor(mx, 1));
    float sum = 0.f;
    for (int c = seg * 256; c < seg * 256 + 256; ++c) {
        float e = expf(s_S[row][c] - mx); s_S[row][c] = e; sum += e;
    }
    sum += __shfl_xor(sum, 1);
    float inv = 1.f / sum;
    for (int c = seg * 256; c < seg * 256 + 256; ++c) s_S[row][c] *= inv;
    __syncthreads();

    v8f acc0 = {}, acc1 = {};
    int r = lane & 15;
    for (int jb = 0; jb < 16; ++jb) {                    // 32 j-cols per WMMA
        v16h ap = pack_fragment(&s_S[r][jb * 32 + kh * 8]);   // ds_load_b128 x4
        v16h bv0 = load_b_tile(V + (long)rowb * CD, CD, jb * 32, hh * 32);
        v16h bv1 = load_b_tile(V + (long)rowb * CD, CD, jb * 32, hh * 32 + 16);
        acc0 = __builtin_amdgcn_wmma_f32_16x16x32_f16(false, ap, false, bv0, (short)0, acc0, false, false);
        acc1 = __builtin_amdgcn_wmma_f32_16x16x32_f16(false, ap, false, bv1, (short)0, acc1, false, false);
    }
    store_c_tile(O, CD, rowb + i0, hh * 32, acc0);
    store_c_tile(O, CD, rowb + i0, hh * 32 + 16, acc1);
}

// h = LN(h + x, g, b)   rowwise over D=128
__global__ void add_ln_kernel(float* __restrict__ h, const float* __restrict__ x,
                              const float* __restrict__ g, const float* __restrict__ b) {
    __shared__ float sred[4];
    int c = threadIdx.x; long row = blockIdx.x;
    float v = h[row * CD + c] + x[row * CD + c];
    float m = block_sum_128(v, sred) * (1.f / CD);
    float d = v - m;
    float var = block_sum_128(d * d, sred) * (1.f / CD);
    h[row * CD + c] = d * rsqrtf(var + 1e-5f) * g[c] + b[c];
}

// out = LN(h + LN(h + sigmoid(gpre)*elu(apre), ng1, nb1), ng2, nb2)
__global__ void gate_fuse_kernel(const float* __restrict__ h, const float* __restrict__ apre,
                                 const float* __restrict__ gpre,
                                 const float* __restrict__ ng1, const float* __restrict__ nb1,
                                 const float* __restrict__ ng2, const float* __restrict__ nb2,
                                 float* __restrict__ out) {
    __shared__ float sred[4];
    int c = threadIdx.x; long row = blockIdx.x;
    float hv = h[row * CD + c];
    float v = hv + sigm_f(gpre[row * CD + c]) * elu_f(apre[row * CD + c]);
    float m = block_sum_128(v, sred) * (1.f / CD);
    float d = v - m;
    float var = block_sum_128(d * d, sred) * (1.f / CD);
    float inner = d * rsqrtf(var + 1e-5f) * ng1[c] + nb1[c];
    float u = hv + inner;
    float m2 = block_sum_128(u, sred) * (1.f / CD);
    float d2 = u - m2;
    float var2 = block_sum_128(d2 * d2, sred) * (1.f / CD);
    out[row * CD + c] = d2 * rsqrtf(var2 + 1e-5f) * ng2[c] + nb2[c];
}

__global__ void ctx_kernel(const float* __restrict__ h, float* __restrict__ ctx) {
    int idx = blockIdx.x * 128 + threadIdx.x;            // B*D = 512
    if (idx >= CB * CD) return;
    int b = idx >> 7, d = idx & (CD - 1);
    float s = 0.f;
    for (int n = 0; n < CN; ++n) s += h[(long)(b * CN + n) * CD + d];
    ctx[idx] = s * (1.f / CN);
}

__global__ void hc_kernel(const float* __restrict__ h, const float* __restrict__ ctx,
                          float* __restrict__ hc) {
    long idx = (long)blockIdx.x * 256 + threadIdx.x;     // rows * 2D
    if (idx >= (long)CB * CN * 2 * CD) return;
    long row = idx >> 8; int c = (int)(idx & 255);
    int b = (int)(row >> 9);
    hc[idx] = (c < CD) ? h[row * CD + c] : ctx[b * CD + (c - CD)];
}

// ---------------------------------------------------------------------------
extern "C" void kernel_launch(void* const* d_in, const int* in_sizes, int n_in,
                              void* d_out, int out_size, void* d_ws, size_t ws_size,
                              hipStream_t stream) {
    const float* node   = (const float*)d_in[0];
    const float* edge   = (const float*)d_in[1];
    const float* t1w    = (const float*)d_in[2];
    const float* t1b    = (const float*)d_in[3];
    const float* t2w    = (const float*)d_in[4];
    const float* t2b    = (const float*)d_in[5];
    const float* taw    = (const float*)d_in[6];
    const float* tng    = (const float*)d_in[7];
    const float* tnb    = (const float*)d_in[8];
    const float* inp_w  = (const float*)d_in[9];
    const float* inp_b  = (const float*)d_in[10];
    const float* te     = (const float*)d_in[11];
    const float* n1w    = (const float*)d_in[12];
    const float* n1b    = (const float*)d_in[13];
    const float* n2w    = (const float*)d_in[14];
    const float* n2b    = (const float*)d_in[15];
    const float* ntg    = (const float*)d_in[16];
    const float* ntb    = (const float*)d_in[17];
    const float* wq     = (const float*)d_in[18];
    const float* wk     = (const float*)d_in[19];
    const float* wv     = (const float*)d_in[20];
    const float* wow    = (const float*)d_in[21];
    const float* wob    = (const float*)d_in[22];
    const float* ebw    = (const float*)d_in[23];
    const float* ebb    = (const float*)d_in[24];
    const float* ang    = (const float*)d_in[25];
    const float* anb    = (const float*)d_in[26];
    const float* ghw    = (const float*)d_in[27];
    const float* ghb    = (const float*)d_in[28];
    const float* ggw    = (const float*)d_in[29];
    const float* ggb    = (const float*)d_in[30];
    const float* gng    = (const float*)d_in[31];
    const float* gnb    = (const float*)d_in[32];
    const float* fng    = (const float*)d_in[33];
    const float* fnb    = (const float*)d_in[34];
    const float* gghw   = (const float*)d_in[35];
    const float* gghb   = (const float*)d_in[36];
    const float* gggw   = (const float*)d_in[37];
    const float* gggb   = (const float*)d_in[38];
    const float* ggng   = (const float*)d_in[39];
    const float* ggnb   = (const float*)d_in[40];
    const float* gnG    = (const float*)d_in[41];
    const float* gnB    = (const float*)d_in[42];

    char* wsp = (char*)d_ws;
    size_t off = 0;
    auto alloc = [&](size_t bytes) -> float* {
        float* p = (float*)(wsp + off);
        off += (bytes + 255) & ~(size_t)255;
        return p;
    };
    const long ROWS = (long)CB * CN;                        // 2048
    float* h0   = alloc((long)CB * CK * CN * CD * 4);       // 8 MB
    float* h    = alloc(ROWS * CD * 4);                     // 1 MB
    float* enc  = alloc((long)CB * CN * CN * CDE * 4);      // 67 MB
    float* bias = alloc((long)CL * CB * CH * CN * CN * 4);  // 33.5 MB
    float* Qb   = alloc(ROWS * CD * 4);
    float* Kb   = alloc(ROWS * CD * 4);
    float* Vb   = alloc(ROWS * CD * 4);
    float* aout = alloc(ROWS * CD * 4);
    float* proj = alloc(ROWS * CD * 4);
    float* apre = alloc(ROWS * CD * 4);
    float* gpre = alloc(ROWS * CD * 4);
    float* ctx  = alloc((long)CB * CD * 4);
    float* hc   = alloc(ROWS * 2 * CD * 4);
    float* ggh  = alloc(ROWS * CD * 4);
    float* ggg  = alloc(ROWS * CD * 4);

    // 1) input projection (WMMA) + time embedding
    wmma_gemm_kernel<<<dim3((CB * CK * CN) / 16, CD / 32), 32, 0, stream>>>(
        node, inp_w, inp_b, h0, CFN, CD);
    add_te_kernel<<<(CB * CK * CN * CD) / 256, 256, 0, stream>>>(h0, te);

    // 2) node temporal conv + LN residual -> h [B,N,D]
    node_conv_kernel<<<CB * CN, CD, 0, stream>>>(h0, n1w, n1b, n2w, n2b, ntg, ntb, h);

    // 3) temporal edge encoder (TDM staging + WMMA convs) -> enc [B,N,N,DE]
    edge_encoder_kernel<<<(CB * CN * CN) / 16, 32, 0, stream>>>(
        edge, t1w, t1b, t2w, t2b, taw, tng, tnb, enc);

    // 4) edge bias for both layers -> bias [L,B,H,N,N]
    edge_bias_kernel<<<(CL * CB * CH * CN * CN) / 256, 256, 0, stream>>>(enc, ebw, ebb, bias);

    // 5) transformer layers
    for (int l = 0; l < CL; ++l) {
        const long mm = (long)l * CD * CD;
        wmma_gemm_kernel<<<dim3(ROWS / 16, CD / 32), 32, 0, stream>>>(h, wq + mm, nullptr, Qb, CD, CD);
        wmma_gemm_kernel<<<dim3(ROWS / 16, CD / 32), 32, 0, stream>>>(h, wk + mm, nullptr, Kb, CD, CD);
        wmma_gemm_kernel<<<dim3(ROWS / 16, CD / 32), 32, 0, stream>>>(h, wv + mm, nullptr, Vb, CD, CD);
        attn_kernel<<<CB * CH * (CN / 16), 32, 0, stream>>>(
            Qb, Kb, Vb, bias + (long)l * CB * CH * CN * CN, aout);
        wmma_gemm_kernel<<<dim3(ROWS / 16, CD / 32), 32, 0, stream>>>(
            aout, wow + mm, wob + l * CD, proj, CD, CD);
        add_ln_kernel<<<ROWS, CD, 0, stream>>>(h, proj, ang + l * CD, anb + l * CD);
        wmma_gemm_kernel<<<dim3(ROWS / 16, CD / 32), 32, 0, stream>>>(
            h, ghw + mm, ghb + l * CD, apre, CD, CD);
        wmma_gemm_kernel<<<dim3(ROWS / 16, CD / 32), 32, 0, stream>>>(
            h, ggw + mm, ggb + l * CD, gpre, CD, CD);
        gate_fuse_kernel<<<ROWS, CD, 0, stream>>>(
            h, apre, gpre, gng + l * CD, gnb + l * CD, fng + l * CD, fnb + l * CD, h);
    }

    // 6) global gating -> d_out
    ctx_kernel<<<(CB * CD + 127) / 128, 128, 0, stream>>>(h, ctx);
    hc_kernel<<<(ROWS * 2 * CD) / 256, 256, 0, stream>>>(h, ctx, hc);
    wmma_gemm_kernel<<<dim3(ROWS / 16, CD / 32), 32, 0, stream>>>(hc, gghw, gghb, ggh, 2 * CD, CD);
    wmma_gemm_kernel<<<dim3(ROWS / 16, CD / 32), 32, 0, stream>>>(hc, gggw, gggb, ggg, 2 * CD, CD);
    gate_fuse_kernel<<<ROWS, CD, 0, stream>>>(h, ggh, ggg, ggng, ggnb, gnG, gnB, (float*)d_out);
}
